// KANLayer_42116449305065
// MI455X (gfx1250) — matrix operand order repaired and checked by reading, not compiled, not run on hardware
//
#include <hip/hip_runtime.h>

typedef _Float16 v16h __attribute__((ext_vector_type(16)));
typedef _Float16 h8   __attribute__((ext_vector_type(8)));
typedef _Float16 h4   __attribute__((ext_vector_type(4)));
typedef float    v8f  __attribute__((ext_vector_type(8)));
typedef float    f4   __attribute__((ext_vector_type(4)));

#define BATCH 2048
#define IN_F  1024
#define OUT_F 1024
#define GRID_G 8
#define BM 128
#define BN 128
#define BK 32
#define LDT 40   // 32 + 8 pad halves -> 80B row stride, conflict-free b128 frag loads

// Fused KAN layer as an augmented-K GEMM on the matrix cores.
//   phase 1 (K=1024): out += x @ base_weight^T
//   phase 2 (K=8192): out += 0.1 * onehot_w(x) @ spline^T  (uniform grid, h=0.25)
// Pipelined: double-buffered LDS, one barrier per K-step, global prefetch into
// registers overlapped with 8 WMMAs per wave per step.
__global__ __launch_bounds__(256) void kan_fused_wmma(
    const float* __restrict__ x,   // (B, I)
    const float* __restrict__ w,   // (O, I)
    const float* __restrict__ s,   // (O, I*G)
    float* __restrict__ out)       // (B, O)
{
  __shared__ _Float16 AsBuf[2][BM * LDT];
  __shared__ _Float16 BsBuf[2][BN * LDT];

  const int tid    = threadIdx.x;
  const int lane   = tid & 31;
  const int wave   = tid >> 5;
  const int blockM = blockIdx.x * BM;
  const int blockN = blockIdx.y * BN;

  // 8 waves in a 4(M) x 2(N) grid; each wave owns a 32x64 patch
  const int mBase = (wave >> 1) * 32;
  const int nBase = (wave & 1) * 64;

  v8f c[2][4];
#pragma unroll
  for (int i = 0; i < 2; ++i)
#pragma unroll
    for (int j = 0; j < 4; ++j) c[i][j] = (v8f){};

  // ---- 128x32 f32 tile -> registers (4 float4 chunks per thread) ----
  auto load_tile = [&](const float* src, int ldsrc, int rowOff, int colOff, f4* v) {
#pragma unroll
    for (int r = 0; r < 4; ++r) {
      int chunk = tid + r * 256;
      int row   = chunk >> 3;          // 8 chunks per 32-wide row
      int kk4   = (chunk & 7) * 4;
      v[r] = *(const f4*)(src + (rowOff + row) * ldsrc + colOff + kk4);
    }
  };
  // ---- registers -> 128x32 f16 LDS tile (scale folded in) ----
  auto store_tile = [&](_Float16* dst, const f4* v, float scale) {
#pragma unroll
    for (int r = 0; r < 4; ++r) {
      int chunk = tid + r * 256;
      int row   = chunk >> 3;
      int kk4   = (chunk & 7) * 4;
      h4 hv;
#pragma unroll
      for (int j = 0; j < 4; ++j) hv[j] = (_Float16)(scale * v[r][j]);
      *(h4*)(dst + row * LDT + kk4) = hv;
    }
  };
  // ---- spline A: one x value per (row, feature) pair, 2 pairs per thread ----
  auto load_splineA = [&](int i0, float* xv) {
#pragma unroll
    for (int r = 0; r < 2; ++r) {
      int p   = tid + r * 256;
      int row = p >> 2;
      int il  = p & 3;
      xv[r] = x[(blockM + row) * IN_F + i0 + il];
    }
  };
  // expand to 8-wide weighted one-hot -> single 16B ds_store_b128
  auto store_splineA = [&](_Float16* dst, const float* xv) {
#pragma unroll
    for (int r = 0; r < 2; ++r) {
      int p   = tid + r * 256;
      int row = p >> 2;
      int il  = p & 3;
      float xc = fminf(fmaxf(xv[r], -1.0f), 1.0f);
      float t  = (xc + 1.0f) * 4.0f;           // uniform grid, h = 0.25
      int  idx = (int)t;                       // t in [0,8], trunc == floor
      idx      = idx > (GRID_G - 1) ? (GRID_G - 1) : idx;
      _Float16 wgt = (_Float16)(t - (float)idx);
      h8 hv;
#pragma unroll
      for (int g = 0; g < GRID_G; ++g) hv[g] = (g == idx) ? wgt : (_Float16)0.0f;
      *(h8*)(dst + row * LDT + il * GRID_G) = hv;
    }
  };

  // ---- fragment loads + 8 WMMAs on staged 128x32 / 128x32 tiles ----
  auto mma_step = [&](const _Float16* As, const _Float16* Bs) {
    // A 16x32 f16 layout (ISA 7.12.2): lane<16 holds K {0..7,16..23} of row lane
    int arow = mBase + (lane & 15);
    int akb  = (lane >> 4) * 8;
    const _Float16* pa0 = &As[arow * LDT + akb];
    const _Float16* pa1 = &As[(arow + 16) * LDT + akb];
    h8 a0lo = *(const h8*)(pa0);
    h8 a0hi = *(const h8*)(pa0 + 16);
    h8 a1lo = *(const h8*)(pa1);
    h8 a1hi = *(const h8*)(pa1 + 16);
    v16h a[2];
#pragma unroll
    for (int j = 0; j < 8; ++j) {
      a[0][j] = a0lo[j]; a[0][j + 8] = a0hi[j];
      a[1][j] = a1lo[j]; a[1][j + 8] = a1hi[j];
    }
    // B 32x16: lane<16 holds K {0..7,16..23} of column lane&15, lane>=16 holds K {8..15,24..31}
    int bcol = nBase + (lane & 15);
    int bkb  = (lane >> 4) * 8;
    v16h b[4];
#pragma unroll
    for (int j = 0; j < 4; ++j) {
      const _Float16* pb = &Bs[(bcol + j * 16) * LDT + bkb];
      h8 blo = *(const h8*)(pb);
      h8 bhi = *(const h8*)(pb + 16);
#pragma unroll
      for (int q = 0; q < 8; ++q) { b[j][q] = blo[q]; b[j][q + 8] = bhi[q]; }
    }
#pragma unroll
    for (int i = 0; i < 2; ++i)
#pragma unroll
      for (int j = 0; j < 4; ++j)
        c[i][j] = __builtin_amdgcn_wmma_f32_16x16x32_f16(
            false, a[i], false, b[j], (short)0, c[i][j], false, false);
  };

  _Float16* curA = AsBuf[0]; _Float16* nxtA = AsBuf[1];
  _Float16* curB = BsBuf[0]; _Float16* nxtB = BsBuf[1];
  f4 av[4], bv[4];
  float xv[2];

  // ---- prologue: stage base tile 0 ----
  load_tile(x, IN_F, blockM, 0, av);
  load_tile(w, IN_F, blockN, 0, bv);
  store_tile(curA, av, 1.0f);
  store_tile(curB, bv, 1.0f);

  // ---- phase 1 steady state: base tiles 0..30, prefetch next base tile ----
  for (int it = 0; it < (IN_F / BK) - 1; ++it) {
    __syncthreads();
    load_tile(x, IN_F, blockM, (it + 1) * BK, av);
    load_tile(w, IN_F, blockN, (it + 1) * BK, bv);
    mma_step(curA, curB);
    store_tile(nxtA, av, 1.0f);
    store_tile(nxtB, bv, 1.0f);
    _Float16* t0 = curA; curA = nxtA; nxtA = t0;
    _Float16* t1 = curB; curB = nxtB; nxtB = t1;
  }
  // ---- phase 1 last tile: prefetch first spline tile ----
  __syncthreads();
  load_splineA(0, xv);
  load_tile(s, IN_F * GRID_G, blockN, 0, bv);
  mma_step(curA, curB);
  store_splineA(nxtA, xv);
  store_tile(nxtB, bv, 0.1f);
  { _Float16* t0 = curA; curA = nxtA; nxtA = t0;
    _Float16* t1 = curB; curB = nxtB; nxtB = t1; }

  // ---- phase 2 steady state: spline tiles 0..254, prefetch next ----
  for (int it = 0; it < (IN_F * GRID_G) / BK - 1; ++it) {
    __syncthreads();
    load_splineA((it + 1) * (BK / GRID_G), xv);
    load_tile(s, IN_F * GRID_G, blockN, (it + 1) * BK, bv);
    mma_step(curA, curB);
    store_splineA(nxtA, xv);
    store_tile(nxtB, bv, 0.1f);
    _Float16* t0 = curA; curA = nxtA; nxtA = t0;
    _Float16* t1 = curB; curB = nxtB; nxtB = t1;
  }
  // ---- epilogue: last spline tile ----
  __syncthreads();
  mma_step(curA, curB);

  // ---- writeback: D 16x16 f32 layout: M = vgpr + 8*(lane>=16), N = lane&15 ----
  int rowLocal = (lane >> 4) * 8;
  int colLocal = lane & 15;
#pragma unroll
  for (int i = 0; i < 2; ++i)
#pragma unroll
    for (int j = 0; j < 4; ++j)
#pragma unroll
      for (int r = 0; r < 8; ++r) {
        int m = blockM + mBase + i * 16 + rowLocal + r;
        int n = blockN + nBase + j * 16 + colLocal;
        out[m * OUT_F + n] = c[i][j][r];
      }
}

extern "C" void kernel_launch(void* const* d_in, const int* in_sizes, int n_in,
                              void* d_out, int out_size, void* d_ws, size_t ws_size,
                              hipStream_t stream) {
  (void)in_sizes; (void)n_in; (void)out_size; (void)d_ws; (void)ws_size;
  const float* x = (const float*)d_in[0];   // (2048, 1024)
  const float* w = (const float*)d_in[1];   // (1024, 1024)
  const float* s = (const float*)d_in[2];   // (1024, 1024, 8)
  // d_in[3] is the grid buffer: fixed linspace(-1,1,9), handled analytically.
  float* out = (float*)d_out;

  dim3 grid(BATCH / BM, OUT_F / BN);        // 16 x 8 blocks
  hipLaunchKernelGGL(kan_fused_wmma, grid, dim3(256), 0, stream, x, w, s, out);
}
